// SFUnit_44169443672622
// MI455X (gfx1250) — matrix operand order repaired and checked by reading, not compiled, not run on hardware
//
#include <hip/hip_runtime.h>
#include <hip/hip_bf16.h>
#include <math.h>

#define Bq 8
#define Nq 64
#define Dm 512
#define Cc 256
#define Hh 128
#define Ww 128
#define Pp 36
#define PGc 128
#define HEADSc 8
#define HDc 64
#define TOTc (Cc*Cc + Pp*Pp)    // 66832
#define NPADc 66880             // TOT padded to multiple of 64
#define Tt (Bq*Nq)              // 512
#define PROW 48                 // P padded to multiple of 16
#define KOUTc (PROW*Cc)         // 12288 padded K for w_out GEMM

typedef __attribute__((ext_vector_type(16))) __bf16 v16bf;
typedef __attribute__((ext_vector_type(8)))  float  v8f;
typedef __attribute__((ext_vector_type(8)))  unsigned short ush8;
typedef __attribute__((ext_vector_type(4)))  unsigned int v4u;
typedef __attribute__((ext_vector_type(8)))  int v8i;
typedef __attribute__((ext_vector_type(4)))  int v4i;

union BFrag { v16bf v; ush8 h[2]; unsigned short s[16]; };

__device__ __forceinline__ unsigned short f2bf(float f) {
  unsigned u = __builtin_bit_cast(unsigned, f);
  unsigned r = u + 0x7FFFu + ((u >> 16) & 1u);  // round-to-nearest-even
  return (unsigned short)(r >> 16);
}
__device__ __forceinline__ float bf2f(unsigned short h) {
  unsigned u = ((unsigned)h) << 16;
  return __builtin_bit_cast(float, u);
}
__device__ __forceinline__ float gelu_f(float x) {
  return 0.5f * x * (1.0f + erff(x * 0.7071067811865476f));
}
__device__ __forceinline__ v8f wmma_bf16(v16bf a, v16bf b, v8f c) {
  return __builtin_amdgcn_wmma_f32_16x16x32_bf16(false, a, false, b, (short)0, c, false, false);
}

// -------- Tensor Data Mover: 2D tile (bf16) global -> LDS, D# per ISA 8.3/8.4 --------
// tile_d0 elements wide (x), tile_d1 rows; row-major packed into LDS.
__device__ __forceinline__ void tdm_load_2d(unsigned lds_addr, const void* gptr,
                                            unsigned tensor_d0, unsigned tensor_d1,
                                            unsigned tile_d0, unsigned tile_d1,
                                            unsigned stride0_elems) {
  unsigned long long ga = (unsigned long long)(size_t)gptr;
  unsigned long long s1 = (unsigned long long)stride0_elems * tensor_d1;   // dim1 stride
  v4u g0;
  g0[0] = 1u;                                            // count=1 valid descriptor
  g0[1] = lds_addr;                                      // LDS byte address
  g0[2] = (unsigned)(ga & 0xffffffffu);                  // global_addr[31:0]
  g0[3] = (unsigned)((ga >> 32) & 0x1ffffffu) | (2u << 30);  // addr[56:32], type=2
  v8i g1;
  g1[0] = (int)(1u << 16);                               // workgroup_mask=0, data_size=1 (2B)
  g1[1] = (int)((tensor_d0 & 0xffffu) << 16);            // tensor_dim0[15:0] at bits 63:48
  g1[2] = (int)((tensor_d0 >> 16) | ((tensor_d1 & 0xffffu) << 16));
  g1[3] = (int)((tensor_d1 >> 16) | (tile_d0 << 16));    // tile_dim0 at bits 127:112
  g1[4] = (int)(tile_d1 & 0xffffu);                      // tile_dim1; tile_dim2=0
  g1[5] = (int)stride0_elems;                            // tensor_dim0_stride[31:0]
  g1[6] = (int)(((unsigned)(s1 & 0xffffu)) << 16);       // stride0 hi16=0 | dim1_stride[15:0]
  g1[7] = (int)(s1 >> 16);                               // dim1_stride[47:16]
  v4i z4 = {0, 0, 0, 0};
  v8i z8 = {0, 0, 0, 0, 0, 0, 0, 0};
  __builtin_amdgcn_tensor_load_to_lds(g0, g1, z4, z4, z8, 0);
}

// Stage B tile (32 x 64 bf16 at (k0, n0), row-major ldb) into LDS column-major
// (col stride 40 ushorts -> 16B aligned columns, banks spread). 128 threads.
__device__ __forceinline__ void stage_b_tile(unsigned short* ldsB,
                                             const unsigned short* __restrict__ Bm,
                                             long ldb, int k0, long n0, int tid) {
  for (int it = 0; it < 2; ++it) {
    int ch = tid + it * 128;            // 256 chunks: r in [0,32), cb in [0,8)
    int r = ch >> 3, cb = ch & 7;
    ush8 v = *(const ush8*)(Bm + (long)(k0 + r) * ldb + n0 + cb * 8);
#pragma unroll
    for (int i = 0; i < 8; ++i) ldsB[(cb * 8 + i) * 40 + r] = v[i];
  }
}

// Read one 32x16 B fragment (cols [16j,16j+16)) from column-major staged LDS tile.
__device__ __forceinline__ v16bf b_frag_lds(const unsigned short* ldsB, int j, int lane) {
  BFrag b;
  int kh = (lane & 16) ? 8 : 0;
  const unsigned short* p = ldsB + (j * 16 + (lane & 15)) * 40 + kh;
  b.h[0] = *(const ush8*)(p);
  b.h[1] = *(const ush8*)(p + 16);
  return b.v;
}

// Read one 32x16 B fragment from a ROW-MAJOR LDS panel (width ldl), rows = K.
__device__ __forceinline__ v16bf b_frag_lds_rm(const unsigned short* ldsB, int ldl,
                                               int k0, int j, int lane) {
  BFrag b;
  int col = j * 16 + (lane & 15);
  int kh = k0 + ((lane & 16) ? 8 : 0);
  const unsigned short* p = ldsB + (long)kh * ldl + col;
#pragma unroll
  for (int i = 0; i < 16; ++i) b.s[i] = p[(i < 8 ? i : i + 8) * ldl];
  return b.v;
}

// A fragment 16x32 at (row base pointer, k0); two 16B loads (ISA 7.12.2 layout).
__device__ __forceinline__ v16bf a_frag(const unsigned short* __restrict__ Arow, int k0) {
  BFrag a;
  a.h[0] = *(const ush8*)(Arow + k0);
  a.h[1] = *(const ush8*)(Arow + k0 + 16);
  return a.v;
}

// ---------------- generic bf16 GEMM: C = act(A@B + bias[n]) (+residual) -------------
// Requires: M%64==0, N%64==0, K%32==0, lda=K, ldb=ldc=N, K and N multiples of 8.
__global__ void gemm_bf16_kernel(const unsigned short* __restrict__ A,
                                 const unsigned short* __restrict__ Bm,
                                 const float* __restrict__ bias, int biasN,
                                 const float* __restrict__ residual,
                                 float* __restrict__ Cf,
                                 unsigned short* __restrict__ Cb,
                                 int M, int N, int K, int act) {
  __shared__ __align__(16) unsigned short ldsB[64 * 40];
  int tid = threadIdx.x;
  int lane = tid & 31, wave = tid >> 5;
  int m0 = blockIdx.y * 64 + wave * 16;
  long n0 = (long)blockIdx.x * 64;
  const unsigned short* Arow = A + (long)(m0 + (lane & 15)) * K + ((lane & 16) ? 8 : 0);
  v8f z = {0.f,0.f,0.f,0.f,0.f,0.f,0.f,0.f};
  v8f acc[4] = {z, z, z, z};
  for (int k0 = 0; k0 < K; k0 += 32) {
    if (k0) __syncthreads();
    stage_b_tile(ldsB, Bm, N, k0, n0, tid);
    __syncthreads();
    v16bf a = a_frag(Arow, k0);
#pragma unroll
    for (int j = 0; j < 4; ++j)
      acc[j] = wmma_bf16(a, b_frag_lds(ldsB, j, lane), acc[j]);
  }
  int ncol = lane & 15;
  int mrow = m0 + ((lane & 16) ? 8 : 0);
#pragma unroll
  for (int j = 0; j < 4; ++j) {
    long n = n0 + 16 * j + ncol;
    int bi = (int)n < biasN ? (int)n : biasN - 1;
    float bv = bias ? bias[bi] : 0.f;
#pragma unroll
    for (int r = 0; r < 8; ++r) {
      long idx = (long)(mrow + r) * N + n;
      float v = acc[j][r] + bv;
      if (residual) v += residual[idx];
      if (act) v = gelu_f(v);
      if (Cb) Cb[idx] = f2bf(v);
      else    Cf[idx] = v;
    }
  }
}

// ------------- einsum 1: h1[t](48x256) = gelu(sampled[t](48x256) @ cm[t](256x256) + m_beta)
// Whole 256x64 cm panel DMA'd to LDS by the Tensor Data Mover (one instruction).
__global__ void einsum_cm_kernel(const unsigned short* __restrict__ sampled,
                                 const unsigned short* __restrict__ params,
                                 const float* __restrict__ m_beta,
                                 unsigned short* __restrict__ h1) {
  __shared__ __align__(16) unsigned short ldsB[256 * 64];   // 32 KB panel
  int t = blockIdx.y;
  int tid = threadIdx.x;
  int lane = tid & 31, wave = tid >> 5;
  int n0 = blockIdx.x * 64;
  const unsigned short* At = sampled + (long)t * PROW * Cc;   // 48 x 256 (rows>=36 junk)
  const unsigned short* Bt = params  + (long)t * NPADc + n0;  // cm panel: 256 rows x 64 cols
  unsigned short* Ht = h1 + (long)t * 64 * Cc;
  if (tid < 32) {                       // wave 0 issues the TDM DMA and waits on TENSORcnt
    tdm_load_2d((unsigned)(size_t)&ldsB[0], Bt,
                /*tensor_d0=*/Cc, /*tensor_d1=*/Cc,
                /*tile_d0=*/64, /*tile_d1=*/Cc, /*stride0=*/Cc);
    __builtin_amdgcn_s_wait_tensorcnt(0);
  }
  __syncthreads();
  if (wave < 3) {
    int m0 = wave * 16;
    const unsigned short* Arow = At + (long)(m0 + (lane & 15)) * Cc + ((lane & 16) ? 8 : 0);
    v8f zz = {0.f,0.f,0.f,0.f,0.f,0.f,0.f,0.f};
    v8f acc[4] = {zz, zz, zz, zz};
    for (int k0 = 0; k0 < Cc; k0 += 32) {
      v16bf a = a_frag(Arow, k0);
#pragma unroll
      for (int j = 0; j < 4; ++j)
        acc[j] = wmma_bf16(a, b_frag_lds_rm(ldsB, 64, k0, j, lane), acc[j]);
    }
    int ncol = lane & 15;
    int mrow = m0 + ((lane & 16) ? 8 : 0);
#pragma unroll
    for (int j = 0; j < 4; ++j) {
      int n = n0 + 16 * j + ncol;
      float bv = m_beta[n];
#pragma unroll
      for (int r = 0; r < 8; ++r)
        Ht[(long)(mrow + r) * Cc + n] = f2bf(gelu_f(acc[j][r] + bv));
    }
  }
}

// ------------- einsum 2: h2[t](48x256) = gelu(smpad[t](48x64) @ h1[t](64x256) + s_beta[row])
// 64x64 h1 panel DMA'd to LDS by the TDM.
__global__ void einsum_sm_kernel(const unsigned short* __restrict__ smpad,
                                 const unsigned short* __restrict__ h1,
                                 const float* __restrict__ s_beta,
                                 unsigned short* __restrict__ h2) {
  __shared__ __align__(16) unsigned short ldsB[64 * 64];    // 8 KB panel
  int t = blockIdx.y;
  int tid = threadIdx.x;
  int lane = tid & 31, wave = tid >> 5;
  int n0 = blockIdx.x * 64;
  const unsigned short* At = smpad + (long)t * PROW * 64;     // 48 x 64, zero beyond 36
  const unsigned short* Bt = h1 + (long)t * 64 * Cc + n0;     // panel: 64 rows x 64 cols
  unsigned short* Ht = h2 + (long)t * PROW * Cc;
  if (tid < 32) {
    tdm_load_2d((unsigned)(size_t)&ldsB[0], Bt,
                /*tensor_d0=*/Cc, /*tensor_d1=*/64,
                /*tile_d0=*/64, /*tile_d1=*/64, /*stride0=*/Cc);
    __builtin_amdgcn_s_wait_tensorcnt(0);
  }
  __syncthreads();
  if (wave < 3) {
    int m0 = wave * 16;
    const unsigned short* Arow = At + (long)(m0 + (lane & 15)) * 64 + ((lane & 16) ? 8 : 0);
    v8f zz = {0.f,0.f,0.f,0.f,0.f,0.f,0.f,0.f};
    v8f acc[4] = {zz, zz, zz, zz};
    for (int k0 = 0; k0 < 64; k0 += 32) {
      v16bf a = a_frag(Arow, k0);
#pragma unroll
      for (int j = 0; j < 4; ++j)
        acc[j] = wmma_bf16(a, b_frag_lds_rm(ldsB, 64, k0, j, lane), acc[j]);
    }
    int ncol = lane & 15;
    int mrow = m0 + ((lane & 16) ? 8 : 0);
#pragma unroll
    for (int j = 0; j < 4; ++j) {
      int n = n0 + 16 * j + ncol;
#pragma unroll
      for (int r = 0; r < 8; ++r) {
        int m = mrow + r;
        float bv = s_beta[m < Pp ? m : Pp - 1];
        Ht[(long)m * Cc + n] = f2bf(gelu_f(acc[j][r] + bv));
      }
    }
  }
}

// ---------------- LayerNorm (D=512), one wave per row, bf16 out ----------------
__global__ void ln_bf_kernel(const float* __restrict__ x, const float* __restrict__ g,
                             const float* __restrict__ b, unsigned short* __restrict__ ybf,
                             int rows) {
  int wave = threadIdx.x >> 5;
  int lane = threadIdx.x & 31;
  int row = blockIdx.x * 4 + wave;
  if (row >= rows) return;
  const float* xr = x + (long)row * Dm;
  float s = 0.f, s2 = 0.f;
  for (int i = lane; i < Dm; i += 32) { float v = xr[i]; s += v; s2 += v * v; }
#pragma unroll
  for (int off = 16; off > 0; off >>= 1) {
    s  += __shfl_xor(s,  off, 32);
    s2 += __shfl_xor(s2, off, 32);
  }
  float mean = s * (1.0f / Dm);
  float var  = s2 * (1.0f / Dm) - mean * mean;
  float inv  = rsqrtf(var + 1e-5f);
  for (int i = lane; i < Dm; i += 32) {
    float v = (xr[i] - mean) * inv * g[i] + b[i];
    ybf[(long)row * Dm + i] = f2bf(v);
  }
}

// ---------------- fp32 -> bf16 elementwise ----------------
__global__ void f32_to_bf16_kernel(const float* __restrict__ x, unsigned short* __restrict__ y, long n) {
  long i = (long)blockIdx.x * blockDim.x + threadIdx.x;
  if (i < n) y[i] = f2bf(x[i]);
}

// ---------------- w_pg2 (PG x TOT) -> bf16 padded (PG x NPAD), zero pad cols ----------------
__global__ void wpg2_pad_kernel(const float* __restrict__ w, unsigned short* __restrict__ out) {
  long i = (long)blockIdx.x * blockDim.x + threadIdx.x;
  const long total = (long)PGc * NPADc;
  if (i >= total) return;
  int col = (int)(i % NPADc);
  int row = (int)(i / NPADc);
  unsigned short v = 0;
  if (col < TOTc) v = f2bf(w[(long)row * TOTc + col]);
  out[i] = v;
}

// ---------------- w_out (9216 x 512) -> bf16 padded (12288 x 512), zero pad rows ----------------
__global__ void wout_pad_kernel(const float* __restrict__ w, unsigned short* __restrict__ out) {
  long i = (long)blockIdx.x * blockDim.x + threadIdx.x;
  const long total = (long)KOUTc * Dm;
  if (i >= total) return;
  int d = (int)(i & (Dm - 1));
  long r = i >> 9;                      // flat (p*256 + c)
  unsigned short v = 0;
  if (r < (long)Pp * Cc) v = f2bf(w[r * Dm + d]);
  out[i] = v;
}

// ---------------- sm part of params -> zero-padded 48 x 64 per token ----------------
__global__ void smpad_kernel(const unsigned short* __restrict__ params,
                             unsigned short* __restrict__ smpad) {
  long i = (long)blockIdx.x * blockDim.x + threadIdx.x;
  const long total = (long)Tt * PROW * 64;
  if (i >= total) return;
  int p = (int)(i & 63);
  long r = i >> 6;
  int q = (int)(r % PROW);
  int t = (int)(r / PROW);
  unsigned short v = 0;
  if (q < Pp && p < Pp) v = params[(long)t * NPADc + Cc * Cc + q * Pp + p];
  smpad[i] = v;
}

// ---------------- img (B,C,H,W) f32 -> (B,H,W,C) bf16 ----------------
__global__ void img_bhwc_kernel(const float* __restrict__ img, unsigned short* __restrict__ out) {
  long i = (long)blockIdx.x * blockDim.x + threadIdx.x;
  const long total = (long)Bq * Hh * Ww * Cc;
  if (i >= total) return;
  int c = (int)(i & (Cc - 1));
  long rest = i >> 8;
  int x = (int)(rest & (Ww - 1)); rest >>= 7;
  int y = (int)(rest & (Hh - 1));
  int b = (int)(rest >> 7);
  out[i] = f2bf(img[(((long)b * Cc + c) * Hh + y) * Ww + x]);
}

// ---------------- offset head: LN_off -> proj(72) -> normalize over P -> grid coords ----------------
__global__ void offset_points_kernel(const float* __restrict__ te, const float* __restrict__ roi,
                                     const float* __restrict__ g, const float* __restrict__ bln,
                                     const float* __restrict__ w_off, const float* __restrict__ bias_off,
                                     float* __restrict__ grid_xy) {
  int t = blockIdx.x;
  int tid = threadIdx.x;                 // 128 threads
  __shared__ float lnv[Dm];
  __shared__ float ps[128], ps2[128];
  __shared__ float offs[2 * Pp];
  __shared__ float stat[4];
  __shared__ float mean_s, inv_s;
  const float* xr = te + (long)t * Dm;
  float s = 0.f, s2 = 0.f;
  for (int i = tid; i < Dm; i += 128) { float v = xr[i]; s += v; s2 += v * v; }
  ps[tid] = s; ps2[tid] = s2;
  __syncthreads();
  if (tid == 0) {
    float a = 0.f, b2 = 0.f;
    for (int i = 0; i < 128; ++i) { a += ps[i]; b2 += ps2[i]; }
    float mean = a * (1.0f / Dm);
    float var  = b2 * (1.0f / Dm) - mean * mean;
    mean_s = mean; inv_s = rsqrtf(var + 1e-5f);
  }
  __syncthreads();
  for (int i = tid; i < Dm; i += 128) lnv[i] = (xr[i] - mean_s) * inv_s * g[i] + bln[i];
  __syncthreads();
  if (tid < 2 * Pp) {
    float acc = bias_off[tid];
    for (int d = 0; d < Dm; ++d) acc += lnv[d] * w_off[(long)d * (2 * Pp) + tid];
    offs[tid] = acc;
  }
  __syncthreads();
  if (tid < 2) {
    float mu = 0.f;
    for (int p = 0; p < Pp; ++p) mu += offs[p * 2 + tid];
    mu *= (1.0f / Pp);
    float v = 0.f;
    for (int p = 0; p < Pp; ++p) { float d = offs[p * 2 + tid] - mu; v += d * d; }
    float sd = sqrtf(v / (Pp - 1)) + 1e-5f;
    stat[tid] = mu; stat[2 + tid] = 3.0f * sd;
  }
  __syncthreads();
  float tlx = roi[(long)t * 4 + 0], tly = roi[(long)t * 4 + 1];
  float brx = roi[(long)t * 4 + 2], bry = roi[(long)t * 4 + 3];
  float cx = (brx + tlx) * 0.5f, cy = (bry + tly) * 0.5f;
  float w  = brx - tlx,          h  = bry - tly;
  if (tid < 2 * Pp) {
    int cidx = tid & 1;
    float on = (offs[tid] - stat[cidx]) / stat[2 + cidx];
    float pt = (cidx == 0 ? cx : cy) + on * (cidx == 0 ? w : h);
    grid_xy[(long)t * (2 * Pp) + tid] = pt * 2.0f - 1.0f;
  }
}

// ---------------- bilinear grid sample from (B,H,W,C) bf16, coalesced over C ----------------
__global__ void grid_sample_kernel(const unsigned short* __restrict__ imgT,
                                   const float* __restrict__ grid_xy,
                                   unsigned short* __restrict__ sampled) {
  int tp = blockIdx.x;                 // t*P + p
  int t = tp / Pp;
  int p = tp - t * Pp;
  int b = t >> 6;
  float gx = grid_xy[(long)tp * 2 + 0];
  float gy = grid_xy[(long)tp * 2 + 1];
  float x = ((gx + 1.0f) * (float)Ww - 1.0f) * 0.5f;
  float y = ((gy + 1.0f) * (float)Hh - 1.0f) * 0.5f;
  float x0f = floorf(x), y0f = floorf(y);
  float wx = x - x0f, wy = y - y0f;
  int x0 = min(max((int)x0f, 0), Ww - 1);
  int x1 = min(max((int)x0f + 1, 0), Ww - 1);
  int y0 = min(max((int)y0f, 0), Hh - 1);
  int y1 = min(max((int)y0f + 1, 0), Hh - 1);
  int c = threadIdx.x;
  const unsigned short* base = imgT + (long)b * Hh * Ww * Cc;
  float v00 = bf2f(base[((long)(y0 * Ww + x0)) * Cc + c]);
  float v01 = bf2f(base[((long)(y0 * Ww + x1)) * Cc + c]);
  float v10 = bf2f(base[((long)(y1 * Ww + x0)) * Cc + c]);
  float v11 = bf2f(base[((long)(y1 * Ww + x1)) * Cc + c]);
  float r = v00 * (1.f - wx) * (1.f - wy) + v01 * wx * (1.f - wy)
          + v10 * (1.f - wx) * wy        + v11 * wx * wy;
  sampled[((long)t * PROW + p) * Cc + c] = f2bf(r);
}

// ---------------- attention: per (b,h) block, N=64, hd=64, fp32, bf16 out ----------------
__global__ void attention_kernel(const float* __restrict__ qkv, unsigned short* __restrict__ obf) {
  int bh = blockIdx.x;
  int b = bh >> 3;
  int h = bh & 7;
  __shared__ float ks[64][64], vs[64][64], att[64][64];
  int tid = threadIdx.x;               // 64 threads; tid = query index
  const float* basep = qkv + ((long)b * Nq) * (3 * Dm) + h * HDc;
  for (int n = 0; n < 64; ++n) {
    ks[n][tid] = basep[(long)n * (3 * Dm) + Dm + tid];
    vs[n][tid] = basep[(long)n * (3 * Dm) + 2 * Dm + tid];
  }
  float qreg[64];
#pragma unroll
  for (int d = 0; d < 64; ++d) qreg[d] = basep[(long)tid * (3 * Dm) + d];
  __syncthreads();
  const float scale = 0.125f;
  float mx = -1e30f;
  for (int k = 0; k < 64; ++k) {
    float sdot = 0.f;
#pragma unroll
    for (int d = 0; d < 64; ++d) sdot += qreg[d] * ks[k][d];
    sdot *= scale;
    att[tid][k] = sdot;
    mx = fmaxf(mx, sdot);
  }
  float sum = 0.f;
  for (int k = 0; k < 64; ++k) {
    float e = __expf(att[tid][k] - mx);
    att[tid][k] = e;
    sum += e;
  }
  float inv = 1.0f / sum;
  for (int d = 0; d < 64; ++d) {
    float o = 0.f;
    for (int k = 0; k < 64; ++k) o += att[tid][k] * vs[k][d];
    obf[((long)(b * Nq + tid)) * Dm + h * HDc + d] = f2bf(o * inv);
  }
}

// =============================== host launcher ===============================
extern "C" void kernel_launch(void* const* d_in, const int* in_sizes, int n_in,
                              void* d_out, int out_size, void* d_ws, size_t ws_size,
                              hipStream_t stream) {
  (void)in_sizes; (void)n_in; (void)out_size; (void)ws_size;
  const float* in_te    = (const float*)d_in[0];
  const float* in_roi   = (const float*)d_in[1];
  const float* in_img   = (const float*)d_in[2];
  const float* ln_off_g = (const float*)d_in[3];
  const float* ln_off_b = (const float*)d_in[4];
  const float* w_off    = (const float*)d_in[5];
  const float* roi_bias = (const float*)d_in[6];
  const float* ln_pg_g  = (const float*)d_in[7];
  const float* ln_pg_b  = (const float*)d_in[8];
  const float* w_pg1    = (const float*)d_in[9];
  const float* b_pg1    = (const float*)d_in[10];
  const float* w_pg2    = (const float*)d_in[11];
  const float* b_pg2    = (const float*)d_in[12];
  const float* m_beta   = (const float*)d_in[13];
  const float* s_beta   = (const float*)d_in[14];
  const float* w_out    = (const float*)d_in[15];
  const float* b_out    = (const float*)d_in[16];
  const float* ln1_g    = (const float*)d_in[17];
  const float* ln1_b    = (const float*)d_in[18];
  const float* w_qkv    = (const float*)d_in[19];
  const float* b_qkv    = (const float*)d_in[20];
  const float* w_proj   = (const float*)d_in[21];
  const float* b_proj   = (const float*)d_in[22];
  const float* ln2_g    = (const float*)d_in[23];
  const float* ln2_b    = (const float*)d_in[24];
  const float* w_fc1    = (const float*)d_in[25];
  const float* b_fc1    = (const float*)d_in[26];
  const float* w_fc2    = (const float*)d_in[27];
  const float* b_fc2    = (const float*)d_in[28];

  char* ws = (char*)d_ws;
  size_t cur = 0;
  auto alloc = [&](size_t bytes) -> void* {
    void* p = ws + cur;
    cur = (cur + bytes + 255) & ~(size_t)255;
    return p;
  };
  float*          te      = (float*)alloc((size_t)Tt * Dm * 4);
  unsigned short* ln_bf   = (unsigned short*)alloc((size_t)Tt * Dm * 2);
  unsigned short* wpg1b   = (unsigned short*)alloc((size_t)Dm * PGc * 2);
  unsigned short* wpg2b   = (unsigned short*)alloc((size_t)PGc * NPADc * 2);
  unsigned short* woutb   = (unsigned short*)alloc((size_t)KOUTc * Dm * 2);
  unsigned short* wqkvb   = (unsigned short*)alloc((size_t)Dm * 3 * Dm * 2);
  unsigned short* wprojb  = (unsigned short*)alloc((size_t)Dm * Dm * 2);
  unsigned short* wfc1b   = (unsigned short*)alloc((size_t)Dm * 4 * Dm * 2);
  unsigned short* wfc2b   = (unsigned short*)alloc((size_t)4 * Dm * Dm * 2);
  unsigned short* pg1b    = (unsigned short*)alloc((size_t)Tt * PGc * 2);
  unsigned short* paramsb = (unsigned short*)alloc((size_t)Tt * NPADc * 2);
  unsigned short* smpadb  = (unsigned short*)alloc((size_t)Tt * PROW * 64 * 2);
  float*          gridxy  = (float*)alloc((size_t)Tt * Pp * 2 * 4);
  unsigned short* imgTb   = (unsigned short*)alloc((size_t)Bq * Hh * Ww * Cc * 2);
  unsigned short* sampb   = (unsigned short*)alloc((size_t)Tt * PROW * Cc * 2);
  unsigned short* h1b     = (unsigned short*)alloc((size_t)Tt * 64 * Cc * 2);
  unsigned short* h2b     = (unsigned short*)alloc((size_t)Tt * PROW * Cc * 2);
  float*          qkvf    = (float*)alloc((size_t)Tt * 3 * Dm * 4);
  unsigned short* ob      = (unsigned short*)alloc((size_t)Tt * Dm * 2);
  unsigned short* fc1b    = (unsigned short*)alloc((size_t)Tt * 4 * Dm * 2);

  (void)hipMemcpyAsync(te, in_te, (size_t)Tt * Dm * 4, hipMemcpyDeviceToDevice, stream);

  auto conv = [&](const float* src, unsigned short* dst, long n) {
    f32_to_bf16_kernel<<<(unsigned)((n + 255) / 256), 256, 0, stream>>>(src, dst, n);
  };
  conv(w_pg1,  wpg1b,  (long)Dm * PGc);
  conv(w_qkv,  wqkvb,  (long)Dm * 3 * Dm);
  conv(w_proj, wprojb, (long)Dm * Dm);
  conv(w_fc1,  wfc1b,  (long)Dm * 4 * Dm);
  conv(w_fc2,  wfc2b,  (long)4 * Dm * Dm);
  {
    long n = (long)PGc * NPADc;
    wpg2_pad_kernel<<<(unsigned)((n + 255) / 256), 256, 0, stream>>>(w_pg2, wpg2b);
  }
  {
    long n = (long)KOUTc * Dm;
    wout_pad_kernel<<<(unsigned)((n + 255) / 256), 256, 0, stream>>>(w_out, woutb);
  }

  auto gemm = [&](const unsigned short* A, const unsigned short* Bm, const float* bias, int biasN,
                  const float* resid, float* Cf, unsigned short* Cb,
                  int M, int N, int K, int act) {
    dim3 g((unsigned)(N / 64), (unsigned)(M / 64));
    gemm_bf16_kernel<<<g, 128, 0, stream>>>(A, Bm, bias, biasN, resid, Cf, Cb, M, N, K, act);
  };

  // --- dynamic-weight parameter path ---
  ln_bf_kernel<<<Tt / 4, 128, 0, stream>>>(te, ln_pg_g, ln_pg_b, ln_bf, Tt);
  gemm(ln_bf, wpg1b, b_pg1, PGc, nullptr, nullptr, pg1b, Tt, PGc, Dm, 0);
  gemm(pg1b, wpg2b, b_pg2, TOTc, nullptr, nullptr, paramsb, Tt, NPADc, PGc, 0);
  {
    long n = (long)Tt * PROW * 64;
    smpad_kernel<<<(unsigned)((n + 255) / 256), 256, 0, stream>>>(paramsb, smpadb);
  }

  // --- sampling path ---
  offset_points_kernel<<<Tt, 128, 0, stream>>>(in_te, in_roi, ln_off_g, ln_off_b,
                                               w_off, roi_bias, gridxy);
  {
    long n = (long)Bq * Hh * Ww * Cc;
    img_bhwc_kernel<<<(unsigned)((n + 255) / 256), 256, 0, stream>>>(in_img, imgTb);
  }
  grid_sample_kernel<<<Tt * Pp, Cc, 0, stream>>>(imgTb, gridxy, sampb);

  // --- per-token dynamic matmuls (batched WMMA, TDM-fed) ---
  einsum_cm_kernel<<<dim3(Cc / 64, Tt), 128, 0, stream>>>(sampb, paramsb, m_beta, h1b);
  einsum_sm_kernel<<<dim3(Cc / 64, Tt), 128, 0, stream>>>(smpadb, h1b, s_beta, h2b);

  // --- output projection + residual ---
  gemm(h2b, woutb, b_out, Dm, te, te, nullptr, Tt, Dm, KOUTc, 0);

  // --- attention block ---
  ln_bf_kernel<<<Tt / 4, 128, 0, stream>>>(te, ln1_g, ln1_b, ln_bf, Tt);
  gemm(ln_bf, wqkvb, b_qkv, 3 * Dm, nullptr, qkvf, nullptr, Tt, 3 * Dm, Dm, 0);
  attention_kernel<<<Bq * HEADSc, 64, 0, stream>>>(qkvf, ob);
  gemm(ob, wprojb, b_proj, Dm, te, te, nullptr, Tt, Dm, Dm, 0);

  // --- MLP block ---
  ln_bf_kernel<<<Tt / 4, 128, 0, stream>>>(te, ln2_g, ln2_b, ln_bf, Tt);
  gemm(ln_bf, wfc1b, b_fc1, 4 * Dm, nullptr, nullptr, fc1b, Tt, 4 * Dm, Dm, 1);
  gemm(fc1b, wfc2b, b_fc2, Dm, te, (float*)d_out, nullptr, Tt, Dm, 4 * Dm, 0);
}
